// MBPerformerEEG_80874234183846
// MI455X (gfx1250) — matrix operand order repaired and checked by reading, not compiled, not run on hardware
//
#include <hip/hip_runtime.h>
#include <math.h>

// ---------------- model dims ----------------
#define BB     16
#define EE     64
#define TTT    1000
#define TP4    250
#define NTOK   256000      // BB * EE * TP4
#define DIMC   32
#define HEADS  4
#define DHH    8
#define NBF    64
#define EPSF   1e-5f

typedef __bf16 bf16_t;
typedef __attribute__((ext_vector_type(16))) __bf16 v16bf;
typedef __attribute__((ext_vector_type(8)))  float  v8f;

enum { MODE_NONE = 0, MODE_GELU = 1 };

__device__ __forceinline__ float gelu_exact(float v) {
    return 0.5f * v * (1.0f + erff(v * 0.70710678118654752f));
}

// =====================================================================
// Tokenizer: band conv (51 taps along T) -> 1x1 conv to 32ch -> BN ->
// exact GELU -> avgpool(1,4) -> h[b, n=e*250+t4, c]
// =====================================================================
__global__ void __launch_bounds__(256)
tokenize_kernel(const float* __restrict__ x, const float* __restrict__ band_w,
                const float* __restrict__ pw_w,
                const float* __restrict__ bn_g, const float* __restrict__ bn_b,
                const float* __restrict__ bn_m, const float* __restrict__ bn_v,
                float* __restrict__ h)
{
    int gid = blockIdx.x * blockDim.x + threadIdx.x;   // [0, 256000)
    if (gid >= BB * EE * TP4) return;
    int t4 = gid % TP4;
    int e  = (gid / TP4) % EE;
    int b  = gid / (TP4 * EE);
    const float* xr = x + ((size_t)(b * EE + e)) * TTT;

    float acc[DIMC];
#pragma unroll
    for (int c = 0; c < DIMC; ++c) acc[c] = 0.0f;

    for (int j = 0; j < 4; ++j) {
        int t = t4 * 4 + j;
        float band[4] = {0.f, 0.f, 0.f, 0.f};
        for (int k = 0; k < 51; ++k) {
            int ti = t + k - 25;
            if (ti >= 0 && ti < TTT) {
                float xv = xr[ti];
#pragma unroll
                for (int o = 0; o < 4; ++o) band[o] += xv * band_w[o * 51 + k];
            }
        }
#pragma unroll
        for (int c = 0; c < DIMC; ++c) {
            float v = band[0] * pw_w[c * 4 + 0] + band[1] * pw_w[c * 4 + 1]
                    + band[2] * pw_w[c * 4 + 2] + band[3] * pw_w[c * 4 + 3];
            v = (v - bn_m[c]) * rsqrtf(bn_v[c] + EPSF) * bn_g[c] + bn_b[c];
            acc[c] += gelu_exact(v);
        }
    }
    float* hp = h + (size_t)gid * DIMC;
#pragma unroll
    for (int c = 0; c < DIMC; ++c) hp[c] = acc[c] * 0.25f;
}

// =====================================================================
// LayerNorm over last dim (32), one thread per token
// =====================================================================
__global__ void __launch_bounds__(256)
layernorm_kernel(const float* __restrict__ src, float* __restrict__ dst,
                 const float* __restrict__ g, const float* __restrict__ bta, int ntok)
{
    int t = blockIdx.x * blockDim.x + threadIdx.x;
    if (t >= ntok) return;
    const float* s = src + (size_t)t * DIMC;
    float mu = 0.f;
#pragma unroll
    for (int c = 0; c < DIMC; ++c) mu += s[c];
    mu *= (1.0f / DIMC);
    float var = 0.f;
#pragma unroll
    for (int c = 0; c < DIMC; ++c) { float d = s[c] - mu; var += d * d; }
    var *= (1.0f / DIMC);
    float r = rsqrtf(var + EPSF);
    float* dp = dst + (size_t)t * DIMC;
#pragma unroll
    for (int c = 0; c < DIMC; ++c) dp[c] = (s[c] - mu) * r * g[c] + bta[c];
}

// =====================================================================
// Generic GEMM via v_wmma_f32_16x16x32_bf16.
// C[M,N] = epilogue( A[M,K] @ W[K,N] + bias (+resid) ), all f32 in memory,
// bf16 inside the matrix pipe, f32 accumulation.
//  - whole W panel staged once per block into LDS as bf16 (<= 6 KB)
//  - A operand fetched as 4x b128 per lane per K-step (two contiguous
//    8-float runs per the ISA 7.12.2 16-bit A layout)
// One wave owns one 16x16 output tile; K marched 32 at a time.
// =====================================================================
#define MAX_KN 3072   // max K*N elements of W across all our GEMMs (32*96)

__global__ void __launch_bounds__(256)
gemm_bf16_wmma(const float* __restrict__ A, int lda,
               const float* __restrict__ W, int ldw,
               const float* __restrict__ bias,
               const float* __restrict__ resid,
               float* __restrict__ C, int ldc,
               int M, int N, int K, int mode)
{
    __shared__ bf16_t s_w[MAX_KN];
    const int tid = threadIdx.x;
    const int kn = K * N;                          // <= MAX_KN by construction
    for (int i = tid; i < kn; i += 256) s_w[i] = (bf16_t)W[i];
    __syncthreads();

    const int wave   = (blockIdx.x * 256 + tid) >> 5;
    const int lane   = tid & 31;
    const int ntiles = N >> 4;
    const int mtiles = M >> 4;
    if (wave >= mtiles * ntiles) return;           // wave-uniform: EXEC stays all-ones
    const int tm   = wave / ntiles;
    const int tn   = wave % ntiles;
    const int r    = lane & 15;
    const int half = lane >> 4;
    const int arow = tm * 16 + r;                  // A-matrix row owned by this lane
    const int col  = tn * 16 + r;                  // B/C/D column owned by this lane

    v8f acc = {0.f, 0.f, 0.f, 0.f, 0.f, 0.f, 0.f, 0.f};

    for (int kb = 0; kb < K; kb += 32) {
        v16bf av, bv;
        // A 16x32 bf16 layout: elem e -> K = kb + half*8 + (e<8 ? e : e+8)
        // => lane reads Ar[0..7] and Ar[16..23]: four b128 loads.
        const float* Ar = A + (size_t)arow * lda + kb + half * 8;
        const float4 a0 = *(const float4*)(Ar + 0);
        const float4 a1 = *(const float4*)(Ar + 4);
        const float4 a2 = *(const float4*)(Ar + 16);
        const float4 a3 = *(const float4*)(Ar + 20);
        av[0]  = (bf16_t)a0.x; av[1]  = (bf16_t)a0.y;
        av[2]  = (bf16_t)a0.z; av[3]  = (bf16_t)a0.w;
        av[4]  = (bf16_t)a1.x; av[5]  = (bf16_t)a1.y;
        av[6]  = (bf16_t)a1.z; av[7]  = (bf16_t)a1.w;
        av[8]  = (bf16_t)a2.x; av[9]  = (bf16_t)a2.y;
        av[10] = (bf16_t)a2.z; av[11] = (bf16_t)a2.w;
        av[12] = (bf16_t)a3.x; av[13] = (bf16_t)a3.y;
        av[14] = (bf16_t)a3.z; av[15] = (bf16_t)a3.w;
        // B 32x16 bf16 layout: elem e -> K = kb + half*8 + (e<8 ? e : e+8), N = col  (from LDS)
        const bf16_t* Wc = s_w + (size_t)(kb + half * 8) * ldw + col;
#pragma unroll
        for (int e = 0; e < 16; ++e) {
            bv[e] = Wc[(size_t)(e < 8 ? e : e + 8) * ldw];
        }
        acc = __builtin_amdgcn_wmma_f32_16x16x32_bf16(
                  false, av, false, bv, (short)0, acc, false, false);
    }

    const float badd = bias ? bias[col] : 0.0f;
#pragma unroll
    for (int i = 0; i < 8; ++i) {
        const int orow = tm * 16 + half * 8 + i;   // D row for VGPR i
        float v = acc[i] + badd;
        if (mode == MODE_GELU) v = gelu_exact(v);
        if (resid) v += resid[(size_t)orow * ldc + col];
        C[(size_t)orow * ldc + col] = v;
    }
}

// =====================================================================
// FAVOR K-features + KV partial accumulation (deterministic, no atomics).
// thread tid = h*64+f owns KV[h,f,:]; per-block partial -> part[block][2048]
// KV[h,f,d] = sum_t (elu(k_h . pm[h,:,f]) + 1) * v[t,h,d]
// =====================================================================
__global__ void __launch_bounds__(256)
kv_accum_kernel(const float* __restrict__ qkv, const float* __restrict__ pm,
                float* __restrict__ part, int ntok, int chunk)
{
    __shared__ float lds[256];   // 4 tokens x (k[32] | v[32])
    const int tid = threadIdx.x;
    const int hh  = tid >> 6;
    const int ff  = tid & 63;

    float pmr[DHH];
#pragma unroll
    for (int d = 0; d < DHH; ++d) pmr[d] = pm[(hh * DHH + d) * NBF + ff];

    float acc[DHH];
#pragma unroll
    for (int d = 0; d < DHH; ++d) acc[d] = 0.f;

    const int t0 = blockIdx.x * chunk;
    for (int tb = 0; tb < chunk; tb += 4) {
        const int g   = tid >> 6;
        const int idx = tid & 63;
        const int t   = t0 + tb + g;
        __syncthreads();
        float v = 0.f;
        if (t < ntok)
            v = (idx < 32) ? qkv[(size_t)t * 96 + 32 + idx]
                           : qkv[(size_t)t * 96 + 64 + (idx - 32)];
        lds[tid] = v;
        __syncthreads();
#pragma unroll
        for (int g2 = 0; g2 < 4; ++g2) {
            const float* kk = &lds[g2 * 64];
            const float* vv = &lds[g2 * 64 + 32];
            float p = 0.f;
#pragma unroll
            for (int d = 0; d < DHH; ++d) p += kk[hh * DHH + d] * pmr[d];
            float kp = (p > 0.f) ? (p + 1.0f) : expf(p);
#pragma unroll
            for (int d = 0; d < DHH; ++d) acc[d] += kp * vv[hh * DHH + d];
        }
    }
    float* pp = part + (size_t)blockIdx.x * 2048 + (size_t)tid * DHH;
#pragma unroll
    for (int d = 0; d < DHH; ++d) pp[d] = acc[d];
}

__global__ void __launch_bounds__(256)
kv_reduce_kernel(const float* __restrict__ part, float* __restrict__ KV, int nblk)
{
    int j = blockIdx.x * blockDim.x + threadIdx.x;   // [0, 2048)
    if (j >= 2048) return;
    float s = 0.f;
    for (int b = 0; b < nblk; ++b) s += part[(size_t)b * 2048 + j];
    KV[j] = s;
}

// =====================================================================
// FAVOR Q-features + apply:  U[d] = sum_f Qp[f]*KV[h,f,d];
// Z = 1/sum_d U[d]  (== 1/(Qp . kvsum));  out[t,h*8+d] = U[d]*Z
// =====================================================================
__global__ void __launch_bounds__(256)
attn_apply_kernel(const float* __restrict__ qkv, const float* __restrict__ pm,
                  const float* __restrict__ KV, float* __restrict__ out, int ntok)
{
    __shared__ float s_pm[HEADS * DHH * NBF];   // 2048
    __shared__ float s_kv[HEADS * NBF * DHH];   // 2048
    const int tid = threadIdx.x;
    for (int i = tid; i < 2048; i += 256) { s_pm[i] = pm[i]; s_kv[i] = KV[i]; }
    __syncthreads();

    const int t = blockIdx.x * 256 + tid;
    if (t >= ntok) return;
    const float* q = qkv + (size_t)t * 96;

    float o[DIMC];
#pragma unroll
    for (int hh = 0; hh < HEADS; ++hh) {
        float xq[DHH];
#pragma unroll
        for (int d = 0; d < DHH; ++d) xq[d] = q[hh * DHH + d];
        float U[DHH];
#pragma unroll
        for (int d = 0; d < DHH; ++d) U[d] = 0.f;
        for (int f = 0; f < NBF; ++f) {
            float p = 0.f;
#pragma unroll
            for (int d = 0; d < DHH; ++d) p += xq[d] * s_pm[(hh * DHH + d) * NBF + f];
            float qp = (p > 0.f) ? (p + 1.0f) : expf(p);
            const float* kvf = &s_kv[(hh * NBF + f) * DHH];
#pragma unroll
            for (int d = 0; d < DHH; ++d) U[d] += qp * kvf[d];
        }
        float den = 0.f;
#pragma unroll
        for (int d = 0; d < DHH; ++d) den += U[d];
        float z = 1.0f / den;
#pragma unroll
        for (int d = 0; d < DHH; ++d) o[hh * DHH + d] = U[d] * z;
    }
    float* op = out + (size_t)t * DIMC;
#pragma unroll
    for (int c = 0; c < DIMC; ++c) op[c] = o[c];
}

// =====================================================================
// Softmax pooling + FC head; one block per batch element
// =====================================================================
__global__ void __launch_bounds__(256)
pool_fc_kernel(const float* __restrict__ y,
               const float* __restrict__ pool_w, const float* __restrict__ pool_b,
               const float* __restrict__ fc_w, const float* __restrict__ fc_b,
               float* __restrict__ out, int npb)
{
    __shared__ float red[256];
    __shared__ float fin[33];
    const int b = blockIdx.x;
    const int tid = threadIdx.x;
    const float pb = pool_b[0];
    float pw[DIMC];
#pragma unroll
    for (int c = 0; c < DIMC; ++c) pw[c] = pool_w[c];
    const float* yb = y + (size_t)b * npb * DIMC;

    float lmax = -3.0e38f;
    for (int t = tid; t < npb; t += 256) {
        const float* yt = yb + (size_t)t * DIMC;
        float l = pb;
#pragma unroll
        for (int c = 0; c < DIMC; ++c) l += yt[c] * pw[c];
        lmax = fmaxf(lmax, l);
    }
    red[tid] = lmax; __syncthreads();
    for (int s = 128; s > 0; s >>= 1) {
        if (tid < s) red[tid] = fmaxf(red[tid], red[tid + s]);
        __syncthreads();
    }
    const float gmax = red[0];
    __syncthreads();

    float lsum = 0.f, lacc[DIMC];
#pragma unroll
    for (int c = 0; c < DIMC; ++c) lacc[c] = 0.f;
    for (int t = tid; t < npb; t += 256) {
        const float* yt = yb + (size_t)t * DIMC;
        float l = pb;
#pragma unroll
        for (int c = 0; c < DIMC; ++c) l += yt[c] * pw[c];
        float e = expf(l - gmax);
        lsum += e;
#pragma unroll
        for (int c = 0; c < DIMC; ++c) lacc[c] += e * yt[c];
    }
    for (int v = 0; v < 33; ++v) {
        red[tid] = (v < 32) ? lacc[v] : lsum;
        __syncthreads();
        for (int s = 128; s > 0; s >>= 1) {
            if (tid < s) red[tid] += red[tid + s];
            __syncthreads();
        }
        if (tid == 0) fin[v] = red[0];
        __syncthreads();
    }
    if (tid == 0) {
        float inv = 1.0f / fin[32];
        for (int cls = 0; cls < 2; ++cls) {
            float o = fc_b[cls];
#pragma unroll
            for (int c = 0; c < DIMC; ++c) o += (fin[c] * inv) * fc_w[c * 2 + cls];
            out[b * 2 + cls] = o;
        }
    }
}

// =====================================================================
// Host orchestration
// =====================================================================
extern "C" void kernel_launch(void* const* d_in, const int* in_sizes, int n_in,
                              void* d_out, int out_size, void* d_ws, size_t ws_size,
                              hipStream_t stream)
{
    (void)in_sizes; (void)n_in; (void)out_size; (void)ws_size;

    const float* x      = (const float*)d_in[0];
    const float* band_w = (const float*)d_in[1];
    const float* pw_w   = (const float*)d_in[2];
    const float* bn_g   = (const float*)d_in[3];
    const float* bn_b   = (const float*)d_in[4];
    const float* bn_m   = (const float*)d_in[5];
    const float* bn_v   = (const float*)d_in[6];
    const float* ln1_g  = (const float*)d_in[7];
    const float* ln1_b  = (const float*)d_in[8];
    const float* wqkv   = (const float*)d_in[9];
    const float* projm  = (const float*)d_in[10];
    const float* wproj  = (const float*)d_in[11];
    const float* bproj  = (const float*)d_in[12];
    const float* ln2_g  = (const float*)d_in[13];
    const float* ln2_b  = (const float*)d_in[14];
    const float* w1     = (const float*)d_in[15];
    const float* b1     = (const float*)d_in[16];
    const float* w2     = (const float*)d_in[17];
    const float* b2     = (const float*)d_in[18];
    const float* norm_g = (const float*)d_in[19];
    const float* norm_b = (const float*)d_in[20];
    const float* pool_w = (const float*)d_in[21];
    const float* pool_b = (const float*)d_in[22];
    const float* fc_w   = (const float*)d_in[23];
    const float* fc_b   = (const float*)d_in[24];
    float* outF = (float*)d_out;

    // Workspace layout (floats): ~166 MB total, fully rewritten each call.
    float* ws     = (float*)d_ws;
    float* h      = ws;                        // NTOK*32  = 8,192,000
    float* y      = ws + 8192000;              // NTOK*32  = 8,192,000
    float* qkv    = ws + 16384000;             // NTOK*96  = 24,576,000 (reused as MLP hidden)
    float* KV     = ws + 40960000;             // 2048
    float* kvpart = ws + 40962048;             // 256*2048 = 524,288

    const int ln_blocks = NTOK / 256;          // 1000

    tokenize_kernel<<<ln_blocks, 256, 0, stream>>>(x, band_w, pw_w,
                                                   bn_g, bn_b, bn_m, bn_v, h);

    for (int l = 0; l < 2; ++l) {
        const float* wqkv_l  = wqkv  + l * DIMC * 96;
        const float* pm_l    = projm + l * HEADS * DHH * NBF;
        const float* wproj_l = wproj + l * DIMC * DIMC;
        const float* bproj_l = bproj + l * DIMC;
        const float* w1_l    = w1    + l * DIMC * 64;
        const float* b1_l    = b1    + l * 64;
        const float* w2_l    = w2    + l * 64 * DIMC;
        const float* b2_l    = b2    + l * DIMC;

        // y = LN1(h)
        layernorm_kernel<<<ln_blocks, 256, 0, stream>>>(h, y, ln1_g + l * DIMC,
                                                        ln1_b + l * DIMC, NTOK);
        // qkv = y @ wqkv  (M=256000, N=96, K=32) -> 96000 tiles / 8 waves
        gemm_bf16_wmma<<<12000, 256, 0, stream>>>(y, DIMC, wqkv_l, 96,
                                                  nullptr, nullptr, qkv, 96,
                                                  NTOK, 96, DIMC, MODE_NONE);
        // KV = sum_{b,n} favorK(k) x v   (deterministic two-stage)
        kv_accum_kernel<<<256, 256, 0, stream>>>(qkv, pm_l, kvpart, NTOK, 1000);
        kv_reduce_kernel<<<8, 256, 0, stream>>>(kvpart, KV, 256);
        // y = attention output per token
        attn_apply_kernel<<<ln_blocks, 256, 0, stream>>>(qkv, pm_l, KV, y, NTOK);
        // h = h + y @ wproj + bproj   (N=32, K=32)
        gemm_bf16_wmma<<<4000, 256, 0, stream>>>(y, DIMC, wproj_l, DIMC,
                                                 bproj_l, h, h, DIMC,
                                                 NTOK, DIMC, DIMC, MODE_NONE);
        // y = LN2(h)
        layernorm_kernel<<<ln_blocks, 256, 0, stream>>>(h, y, ln2_g + l * DIMC,
                                                        ln2_b + l * DIMC, NTOK);
        // qkv(buf) = gelu(y @ w1 + b1)   (N=64, K=32)
        gemm_bf16_wmma<<<8000, 256, 0, stream>>>(y, DIMC, w1_l, 64,
                                                 b1_l, nullptr, qkv, 64,
                                                 NTOK, 64, DIMC, MODE_GELU);
        // h = h + qkv(buf) @ w2 + b2     (N=32, K=64)
        gemm_bf16_wmma<<<4000, 256, 0, stream>>>(qkv, 64, w2_l, DIMC,
                                                 b2_l, h, h, DIMC,
                                                 NTOK, DIMC, 64, MODE_NONE);
    }

    // y = LN_final(h); softmax-pool + FC -> out[16,2]
    layernorm_kernel<<<ln_blocks, 256, 0, stream>>>(h, y, norm_g, norm_b, NTOK);
    pool_fc_kernel<<<BB, 256, 0, stream>>>(y, pool_w, pool_b, fc_w, fc_b,
                                           outF, EE * TP4);
}
